// HiSACKT_11046655885796
// MI455X (gfx1250) — matrix operand order repaired and checked by reading, not compiled, not run on hardware
//
#include <hip/hip_runtime.h>
#include <hip/hip_bf16.h>
#include <cstdint>

// ---------------------------------------------------------------------------
// Types for CDNA5 WMMA (wave32, v_wmma_f32_16x16x32_bf16)
// ---------------------------------------------------------------------------
typedef __attribute__((ext_vector_type(16))) __bf16 v16bf;
typedef __attribute__((ext_vector_type(8)))  __bf16 v8bf;
typedef __attribute__((ext_vector_type(8)))  float  v8f;

#define DEV static __device__ __forceinline__

DEV unsigned short f2bf(float f) {
  unsigned int u = __float_as_uint(f);
  unsigned int r = u + 0x7FFFu + ((u >> 16) & 1u);   // round-to-nearest-even
  return (unsigned short)(r >> 16);
}
DEV float bf2f(unsigned short h) { return __uint_as_float(((unsigned int)h) << 16); }

constexpr int kNumQ = 2048;
constexpr int kN    = 512;
constexpr int kD    = 256;
constexpr int kH    = 8;
constexpr int kLW   = 25;
constexpr int kB    = 32;
constexpr long kBN  = (long)kB * kN;

typedef unsigned short u16;

// LDS byte-offset of a pointer into static __shared__ storage (addrspace(3)
// pointers are 32-bit on amdgcn — exactly the VDST operand of async LDS loads).
DEV unsigned ldsoff(const void* p) {
  return (unsigned)(unsigned long long)(__attribute__((address_space(3))) const void*)p;
}

// CDNA5 async global->LDS copy, 16 bytes per lane, tracked by ASYNCcnt.
DEV void async_ld16(unsigned lds, const void* g) {
  asm volatile("global_load_async_to_lds_b128 %0, %1, off"
               :: "v"(lds), "v"(g) : "memory");
}

// build a v16bf fragment from two 16B chunks
DEV v16bf frag(const u16* lo_p, const u16* hi_p) {
  v8bf lo = *reinterpret_cast<const v8bf*>(lo_p);
  v8bf hi = *reinterpret_cast<const v8bf*>(hi_p);
  v16bf r;
#pragma unroll
  for (int i = 0; i < 8; i++) { r[i] = lo[i]; r[i + 8] = hi[i]; }
  return r;
}

// ---------------------------------------------------------------------------
// Elementwise helpers
// ---------------------------------------------------------------------------
__global__ void k_f2bf(const float* __restrict__ s, u16* __restrict__ d, long n) {
  long i = (long)blockIdx.x * blockDim.x + threadIdx.x;
  if (i < n) d[i] = f2bf(s[i]);
}

__global__ void k_zero16(u16* __restrict__ d, long n) {
  long i = (long)blockIdx.x * blockDim.x + threadIdx.x;
  if (i < n) d[i] = 0;
}

// M = M_emb[q + NUM_Q*r] + P ; E = E_emb[qry]  (f32 + bf16 copies)
__global__ void k_embed(const int* __restrict__ qx, const int* __restrict__ rx,
                        const int* __restrict__ qy,
                        const float* __restrict__ Memb, const float* __restrict__ Eemb,
                        const float* __restrict__ Pemb,
                        float* __restrict__ Mf, float* __restrict__ Ef,
                        u16* __restrict__ Mbf, u16* __restrict__ Ebf) {
  long gid = (long)blockIdx.x * blockDim.x + threadIdx.x;
  if (gid >= kBN * kD) return;
  int d = (int)(gid % kD);
  long i = gid / kD;
  int n = (int)(i % kN);
  int x = qx[i] + kNumQ * rx[i];
  float m = Memb[(long)x * kD + d] + Pemb[(long)n * kD + d];
  float e = Eemb[(long)qy[i] * kD + d];
  Mf[gid] = m; Ef[gid] = e;
  Mbf[gid] = f2bf(m); Ebf[gid] = f2bf(e);
}

// stable row softmax -> bf16 (one block per row)
__global__ void k_rowsoftmax_bf(const float* __restrict__ src, u16* __restrict__ dst, int cols) {
  int row = blockIdx.x, t = threadIdx.x;
  const float* sr = src + (long)row * cols;
  __shared__ float red[256];
  float mx = -3.4e38f;
  for (int k = t; k < cols; k += 256) mx = fmaxf(mx, sr[k]);
  red[t] = mx; __syncthreads();
  for (int o = 128; o; o >>= 1) { if (t < o) red[t] = fmaxf(red[t], red[t + o]); __syncthreads(); }
  mx = red[0]; __syncthreads();
  float s = 0.f;
  for (int k = t; k < cols; k += 256) s += __expf(sr[k] - mx);
  red[t] = s; __syncthreads();
  for (int o = 128; o; o >>= 1) { if (t < o) red[t] += red[t + o]; __syncthreads(); }
  float inv = 1.f / red[0];
  u16* dr = dst + (long)row * cols;
  for (int k = t; k < cols; k += 256) dr[k] = f2bf(__expf(sr[k] - mx) * inv);
}

// layernorm(a + b [+ c]) -> f32 (+ optional bf16). one block (256 thr) per row, D==256.
__global__ void k_add3_ln(const float* __restrict__ a, const float* __restrict__ b,
                          const float* __restrict__ c,
                          const float* __restrict__ g, const float* __restrict__ beta,
                          float* __restrict__ outf, u16* __restrict__ outbf) {
  int row = blockIdx.x, d = threadIdx.x;
  __shared__ float red[256];
  long idx = (long)row * kD + d;
  float v = a[idx] + b[idx] + (c ? c[idx] : 0.f);
  red[d] = v; __syncthreads();
  for (int o = 128; o; o >>= 1) { if (d < o) red[d] += red[d + o]; __syncthreads(); }
  float mean = red[0] * (1.f / kD); __syncthreads();
  float dv = v - mean;
  red[d] = dv * dv; __syncthreads();
  for (int o = 128; o; o >>= 1) { if (d < o) red[d] += red[d + o]; __syncthreads(); }
  float var = red[0] * (1.f / kD);
  float o = dv * rsqrtf(var + 1e-5f) * g[d] + beta[d];
  outf[idx] = o;
  if (outbf) outbf[idx] = f2bf(o);
}

// pack last LW rows of S_base (bf16) per batch into contiguous (B*LW, D)
__global__ void k_gather_loc(const u16* __restrict__ Sb, u16* __restrict__ loc) {
  long gid = (long)blockIdx.x * blockDim.x + threadIdx.x;
  long total = (long)kB * kLW * kD;
  if (gid >= total) return;
  int d = (int)(gid % kD);
  long i = gid / kD;
  int b = (int)(i / kLW), j = (int)(i % kLW);
  loc[gid] = Sb[((long)(b * kN + (kN - kLW) + j)) * kD + d];
}

// scatter local attention output (bf16) into zero-padded full (B,N,D)
__global__ void k_scatter_loc(const u16* __restrict__ locout, u16* __restrict__ full) {
  long gid = (long)blockIdx.x * blockDim.x + threadIdx.x;
  long total = (long)kB * kLW * kD;
  if (gid >= total) return;
  int d = (int)(gid % kD);
  long i = gid / kD;
  int b = (int)(i / kLW), j = (int)(i % kLW);
  full[((long)(b * kN + (kN - kLW) + j)) * kD + d] = locout[gid];
}

// SGR[b,n,:] = SWP[q[b,n], :]   (SWP already has projection bias folded in)
__global__ void k_gather_sgr(const float* __restrict__ SWP, const int* __restrict__ qx,
                             u16* __restrict__ SGR) {
  long gid = (long)blockIdx.x * blockDim.x + threadIdx.x;
  if (gid >= kBN * kD) return;
  int d = (int)(gid % kD);
  long i = gid / kD;
  SGR[gid] = f2bf(SWP[(long)qx[i] * kD + d]);
}

// S = Sbase + Ssg + Sglob + Sbl + Ssgl + (local rows)
__global__ void k_final_sum(const float* __restrict__ sb, const float* __restrict__ ssg,
                            const float* __restrict__ sg, const float* __restrict__ sbl,
                            const float* __restrict__ ssgl, const float* __restrict__ sloc,
                            float* __restrict__ outf, u16* __restrict__ outbf) {
  long gid = (long)blockIdx.x * blockDim.x + threadIdx.x;
  if (gid >= kBN * kD) return;
  int d = (int)(gid % kD);
  long i = gid / kD;
  int n = (int)(i % kN);
  int b = (int)(i / kN);
  float s = sb[gid] + ssg[gid] + sg[gid] + sbl[gid] + ssgl[gid];
  int j = n - (kN - kLW);
  if (j >= 0) s += sloc[((long)(b * kLW + j)) * kD + d];
  outf[gid] = s;
  outbf[gid] = f2bf(s);
}

// p = sigmoid(F @ pred_w^T + pred_b), one wave per row
__global__ void k_pred(const float* __restrict__ F, const float* __restrict__ pw,
                       const float* __restrict__ pb, float* __restrict__ out, int rows) {
  int row = blockIdx.x * 8 + (threadIdx.x >> 5);
  int lane = threadIdx.x & 31;
  if (row >= rows) return;
  float s = 0.f;
  for (int d = lane; d < kD; d += 32) s += F[(long)row * kD + d] * pw[d];
  for (int o = 16; o; o >>= 1) s += __shfl_down(s, o);
  if (lane == 0) out[row] = 1.f / (1.f + __expf(-(s + pb[0])));
}

// V (B*Lk, D) bf16 -> Vt (B, D, Lkp) bf16, zero padded keys
__global__ void k_vt(const u16* __restrict__ V, u16* __restrict__ Vt, int Lk, int Lkp) {
  long gid = (long)blockIdx.x * blockDim.x + threadIdx.x;
  long total = (long)kB * kD * Lkp;
  if (gid >= total) return;
  int k = (int)(gid % Lkp);
  long rest = gid / Lkp;
  int d = (int)(rest % kD);
  int b = (int)(rest / kD);
  Vt[gid] = (k < Lk) ? V[((long)(b * Lk + k)) * kD + d] : (u16)0;
}

// ---------------------------------------------------------------------------
// Generic batched WMMA GEMM:  C[M,N] = A[M,K] * W[N,K]^T + bias
// bf16 in, fp32 accumulate, optional f32/bf16 out, optional ReLU.
// Block = 256 threads = 8 waves (2x4), each wave a 32x32 tile (2x2 WMMA),
// block tile = 64x128. K-tiles (64x32 A, 128x32 B) are double-buffered in LDS
// and filled with CDNA5 async global->LDS loads (ASYNCcnt pipeline).
// Requires N%128==0, K%32==0; M handled with clamp+guard.
// ---------------------------------------------------------------------------
__global__ __launch_bounds__(256) void k_gemm(
    const u16* __restrict__ A, const u16* __restrict__ W,
    const float* __restrict__ bias, float* __restrict__ Cf, u16* __restrict__ Cbf,
    int M, int N, int K, int relu, long Ab, long Wb, long Cb) {
  __shared__ __align__(16) u16 As[2][64 * 32];
  __shared__ __align__(16) u16 Bs[2][128 * 32];
  A += (long)blockIdx.z * Ab;
  W += (long)blockIdx.z * Wb;
  const long cbase = (long)blockIdx.z * Cb;
  const int tid = threadIdx.x, wave = tid >> 5, lane = tid & 31;
  const int sel = lane >> 4, lm = lane & 15;
  const int wr = wave >> 2;           // wave row 0..1
  const int wc = wave & 3;            // wave col 0..3
  const int row0 = blockIdx.x * 64;
  const int col0 = blockIdx.y * 128;
  // staging assignment: A 64 rows x 4 16B-chunks (1/thread), B 128 cols x 4 (2/thread)
  const int ar = tid >> 2, apart = tid & 3;
  const int bc = tid >> 1, bpart = (tid & 1) * 2;
  int arowg = row0 + ar; if (arowg >= M) arowg = M - 1;
  const u16* agp = A + (long)arowg * K + apart * 8;
  const u16* bgp = W + (long)(col0 + bc) * K + bpart * 8;

  auto stage = [&](int buf, int kb) {
    async_ld16(ldsoff(&As[buf][ar * 32 + apart * 8]), agp + kb);
    async_ld16(ldsoff(&Bs[buf][bc * 32 + bpart * 8]), bgp + kb);
    async_ld16(ldsoff(&Bs[buf][bc * 32 + (bpart + 1) * 8]), bgp + kb + 8);
  };

  v8f acc00 = {}, acc01 = {}, acc10 = {}, acc11 = {};
  const int nk = K / 32;
  stage(0, 0);
  for (int i = 0; i < nk; ++i) {
    const int buf = i & 1;
    __syncthreads();  // all waves done reading the buffer we are about to refill
    if (i + 1 < nk) {
      stage(buf ^ 1, (i + 1) * 32);
      asm volatile("s_wait_asynccnt 0x3" ::: "memory");  // tile i landed (in-order)
    } else {
      asm volatile("s_wait_asynccnt 0x0" ::: "memory");
    }
    __syncthreads();  // tile i visible to the whole block
    const u16* a0p = &As[buf][(wr * 32 + lm) * 32];
    const u16* a1p = &As[buf][(wr * 32 + 16 + lm) * 32];
    const u16* b0p = &Bs[buf][(wc * 32 + lm) * 32];
    const u16* b1p = &Bs[buf][(wc * 32 + 16 + lm) * 32];
    v16bf a0 = frag(a0p + sel * 16, a0p + sel * 16 + 8);
    v16bf a1 = frag(a1p + sel * 16, a1p + sel * 16 + 8);
    v16bf b0 = frag(b0p + sel * 16, b0p + sel * 16 + 8);
    v16bf b1 = frag(b1p + sel * 16, b1p + sel * 16 + 8);
    acc00 = __builtin_amdgcn_wmma_f32_16x16x32_bf16(false, a0, false, b0, (short)0, acc00, false, false);
    acc01 = __builtin_amdgcn_wmma_f32_16x16x32_bf16(false, a0, false, b1, (short)0, acc01, false, false);
    acc10 = __builtin_amdgcn_wmma_f32_16x16x32_bf16(false, a1, false, b0, (short)0, acc10, false, false);
    acc11 = __builtin_amdgcn_wmma_f32_16x16x32_bf16(false, a1, false, b1, (short)0, acc11, false, false);
  }
#pragma unroll
  for (int t = 0; t < 2; t++) {
#pragma unroll
    for (int u = 0; u < 2; u++) {
      const v8f& acc = t ? (u ? acc11 : acc10) : (u ? acc01 : acc00);
#pragma unroll
      for (int rr = 0; rr < 8; rr++) {
        int m = rr + sel * 8;
        int grow = row0 + wr * 32 + t * 16 + m;
        int gcol = col0 + wc * 32 + u * 16 + lm;
        if (grow < M) {
          float v = acc[rr];
          if (bias) v += bias[gcol];
          if (relu) v = fmaxf(v, 0.f);
          long idx = cbase + (long)grow * N + gcol;
          if (Cf)  Cf[idx]  = v;
          if (Cbf) Cbf[idx] = f2bf(v);
        }
      }
    }
  }
}

// ---------------------------------------------------------------------------
// Fused multi-head attention core. One block (128 thr = 4 waves) per
// (16-query tile, batch); loops over all 8 heads. QK^T and P*V use WMMA,
// softmax in fp32 in LDS. Optionally accumulates head-mean weights -> wout.
// ---------------------------------------------------------------------------
__global__ __launch_bounds__(128) void k_attn(
    const u16* __restrict__ Q, const u16* __restrict__ K, const u16* __restrict__ Vt,
    u16* __restrict__ O, float* __restrict__ wout,
    int Lq, int Lk, int Lkp, int causal, float scale) {
  extern __shared__ char smem[];
  float* sc   = (float*)smem;                         // 16*512 f32
  u16*   pbf  = (u16*)(smem + 16 * 512 * 4);          // 16*512 bf16
  float* oacc = (float*)(smem + 16 * 512 * 4 + 16 * 512 * 2);          // 4*16*32 f32
  float* wacc = (float*)(smem + 16 * 512 * 4 + 16 * 512 * 2 + 4 * 16 * 32 * 4);
  const int tid = threadIdx.x, wave = tid >> 5, lane = tid & 31;
  const int sel = lane >> 4, lm = lane & 15;
  const int b = blockIdx.y;
  const int q0 = blockIdx.x * 16;
  const long qbase = (long)b * Lq;
  const long kbase = (long)b * Lk;
  const long vtbase = (long)b * kD * Lkp;
  const int nkt = Lkp >> 4;

  if (wout) for (int i = tid; i < 16 * 512; i += 128) wacc[i] = 0.f;

  for (int h = 0; h < kH; ++h) {
    int qrow = q0 + lm; if (qrow >= Lq) qrow = Lq - 1;
    const u16* qp = Q + (qbase + qrow) * kD + h * 32;
    v16bf qa = frag(qp + sel * 16, qp + sel * 16 + 8);
    // ---- scores: Q (16x32) x K^T per 16-key tile ----
    for (int kt = wave; kt < nkt; kt += 4) {
      int key = kt * 16 + lm;
      int keyc = key < Lk ? key : (Lk - 1);
      const u16* kp = K + (kbase + keyc) * kD + h * 32 + sel * 16;
      v16bf kf = frag(kp, kp + 8);
      v8f c = {};
      c = __builtin_amdgcn_wmma_f32_16x16x32_bf16(false, qa, false, kf, (short)0, c,
                                                  false, false);
#pragma unroll
      for (int rr = 0; rr < 8; rr++) {
        int m = rr + sel * 8;
        int kk = kt * 16 + lm;
        int qg = q0 + m;
        bool msk = (kk >= Lk) || (causal && kk > qg);
        sc[m * 512 + kk] = msk ? -1e9f : c[rr] * scale;
      }
    }
    __syncthreads();
    // ---- softmax: 8 threads per row ----
    {
      int row = tid >> 3, sub = tid & 7;
      float mx = -3.4e38f;
      for (int k = sub; k < Lkp; k += 8) mx = fmaxf(mx, sc[row * 512 + k]);
      mx = fmaxf(mx, __shfl_xor(mx, 1));
      mx = fmaxf(mx, __shfl_xor(mx, 2));
      mx = fmaxf(mx, __shfl_xor(mx, 4));
      float sum = 0.f;
      for (int k = sub; k < Lkp; k += 8) {
        float e = __expf(sc[row * 512 + k] - mx);
        sc[row * 512 + k] = e;
        sum += e;
      }
      sum += __shfl_xor(sum, 1);
      sum += __shfl_xor(sum, 2);
      sum += __shfl_xor(sum, 4);
      float inv = 1.f / sum;
      for (int k = sub; k < Lkp; k += 8) {
        float p = sc[row * 512 + k] * inv;
        pbf[row * 512 + k] = f2bf(p);
        if (wout) wacc[row * 512 + k] += p;
      }
    }
    __syncthreads();
    // ---- P (16xLkp) x V (Lkp x 32): each wave a K-slice, reduce in LDS ----
    v8f a0 = {}, a1 = {};
    for (int kb2 = wave * 32; kb2 < Lkp; kb2 += 128) {
      const u16* pp = pbf + lm * 512 + kb2;
      v16bf pa = frag(pp + sel * 16, pp + sel * 16 + 8);
      const u16* vp0 = Vt + vtbase + (long)(h * 32 + lm) * Lkp + kb2 + sel * 16;
      const u16* vp1 = Vt + vtbase + (long)(h * 32 + 16 + lm) * Lkp + kb2 + sel * 16;
      v16bf vb0 = frag(vp0, vp0 + 8);
      v16bf vb1 = frag(vp1, vp1 + 8);
      a0 = __builtin_amdgcn_wmma_f32_16x16x32_bf16(false, pa, false, vb0, (short)0, a0,
                                                   false, false);
      a1 = __builtin_amdgcn_wmma_f32_16x16x32_bf16(false, pa, false, vb1, (short)0, a1,
                                                   false, false);
    }
#pragma unroll
    for (int rr = 0; rr < 8; rr++) {
      int m = rr + sel * 8;
      oacc[(wave * 16 + m) * 32 + lm] = a0[rr];
      oacc[(wave * 16 + m) * 32 + 16 + lm] = a1[rr];
    }
    __syncthreads();
    for (int e = tid; e < 16 * 32; e += 128) {
      int m = e >> 5, d = e & 31;
      float s = oacc[m * 32 + d] + oacc[(16 + m) * 32 + d] +
                oacc[(32 + m) * 32 + d] + oacc[(48 + m) * 32 + d];
      int qg = q0 + m;
      if (qg < Lq) O[(qbase + qg) * kD + h * 32 + d] = f2bf(s);
    }
    __syncthreads();
  }
  if (wout) {
    for (long e = tid; e < (long)16 * Lk; e += 128) {
      int m = (int)(e / Lk), k = (int)(e % Lk);
      int qg = q0 + m;
      if (qg < Lq)
        wout[(long)b * Lq * Lk + (long)qg * Lk + k] = wacc[m * 512 + k] * (1.f / kH);
    }
  }
}

// ---------------------------------------------------------------------------
// Host orchestration
// ---------------------------------------------------------------------------
extern "C" void kernel_launch(void* const* d_in, const int* in_sizes, int n_in,
                              void* d_out, int out_size, void* d_ws, size_t ws_size,
                              hipStream_t stream) {
  (void)in_sizes; (void)n_in; (void)out_size; (void)ws_size;
  const int*   qi   = (const int*)d_in[0];
  const int*   ri   = (const int*)d_in[1];
  const int*   qryi = (const int*)d_in[2];
  const float* Memb = (const float*)d_in[3];
  const float* Eemb = (const float*)d_in[4];
  const float* Pemb = (const float*)d_in[5];
  const float* w_in = (const float*)d_in[6];
  const float* b_in = (const float*)d_in[7];
  const float* w_out= (const float*)d_in[8];
  const float* b_out= (const float*)d_in[9];
  const float* scw  = (const float*)d_in[10];
  const float* spw  = (const float*)d_in[11];
  const float* spb  = (const float*)d_in[12];
  const float* stew = (const float*)d_in[13];
  const float* steb = (const float*)d_in[14];
  const float* f1w  = (const float*)d_in[15];
  const float* f1b  = (const float*)d_in[16];
  const float* f2w  = (const float*)d_in[17];
  const float* f2b  = (const float*)d_in[18];
  const float* ln1g = (const float*)d_in[19];
  const float* ln1b = (const float*)d_in[20];
  const float* ln2g = (const float*)d_in[21];
  const float* ln2b = (const float*)d_in[22];
  const float* pw   = (const float*)d_in[23];
  const float* pb   = (const float*)d_in[24];
  float* out = (float*)d_out;

  char* base = (char*)d_ws;
  size_t off = 0;
  auto alloc = [&](size_t bytes) -> char* {
    off = (off + 255) & ~(size_t)255;
    char* p = base + off;
    off += bytes;
    return p;
  };

  const size_t act_f = (size_t)kBN * kD * 4;
  const size_t act_b = (size_t)kBN * kD * 2;

  // bf16 weights
  u16* w_in_bf  = (u16*)alloc((size_t)6 * 3 * kD * kD * 2);
  u16* w_out_bf = (u16*)alloc((size_t)6 * kD * kD * 2);
  u16* spw_bf   = (u16*)alloc((size_t)kD * kNumQ * 2);
  u16* stew_bf  = (u16*)alloc((size_t)kD * kN * 2);
  u16* f1w_bf   = (u16*)alloc((size_t)kD * kD * 2);
  u16* f2w_bf   = (u16*)alloc((size_t)kD * kD * 2);
  u16* sw_bf    = (u16*)alloc((size_t)kNumQ * kNumQ * 2);
  float* SWPf   = (float*)alloc((size_t)kNumQ * kD * 4);

  // activations f32
  float* Mf     = (float*)alloc(act_f);
  float* Ef     = (float*)alloc(act_f);
  float* S0f    = (float*)alloc(act_f);
  float* Sbasef = (float*)alloc(act_f);
  float* Ssgf   = (float*)alloc(act_f);
  float* Sglobf = (float*)alloc(act_f);
  float* Sblf   = (float*)alloc(act_f);
  float* Ssglf  = (float*)alloc(act_f);
  float* Slocout_f = (float*)alloc((size_t)kB * kLW * kD * 4);
  // activations bf16
  u16* Mbf         = (u16*)alloc(act_b);
  u16* Ebf         = (u16*)alloc(act_b);
  u16* Sbase_bf    = (u16*)alloc(act_b);
  u16* Slocfull_bf = (u16*)alloc(act_b);
  u16* SGR_bf      = (u16*)alloc(act_b);
  u16* Ssg0_bf     = (u16*)alloc(act_b);
  u16* Ssg_bf      = (u16*)alloc(act_b);
  u16* Sglob_bf    = (u16*)alloc(act_b);
  u16* Ssum_bf     = (u16*)alloc(act_b);
  u16* H1_bf       = (u16*)alloc(act_b);
  u16* G_bf        = (u16*)alloc((size_t)kB * kN * kN * 2);
  u16* loc_bf      = (u16*)alloc((size_t)kB * kLW * kD * 2);
  u16* Slocout_bf  = (u16*)alloc((size_t)kB * kLW * kD * 2);
  // MHA scratch (reused)
  u16* Qbf   = (u16*)alloc(act_b);
  u16* Kbf   = (u16*)alloc(act_b);
  u16* Vbf   = (u16*)alloc(act_b);
  u16* Obf   = (u16*)alloc(act_b);
  u16* Vt_bf = (u16*)alloc((size_t)kB * kD * kN * 2);
  // reuse dead f32 buffers
  float* Ssumf = Ef;   // E dead after ln1
  float* Ff    = S0f;  // S0 dead after ln1
  float* F2f   = Mf;   // M dead after ln1

  auto gl = [&](long n) { return dim3((unsigned)((n + 255) / 256)); };

  auto gemm = [&](const u16* A, const u16* W, const float* bias, float* Cf, u16* Cbf,
                  int M, int N, int K, int relu, int batches, long Ab, long Wb, long Cb) {
    dim3 g((unsigned)((M + 63) / 64), (unsigned)(N / 128), (unsigned)batches);
    k_gemm<<<g, 256, 0, stream>>>(A, W, bias, Cf, Cbf, M, N, K, relu, Ab, Wb, Cb);
  };

  auto mha = [&](const u16* Xq, int Lq, const u16* Xkv, int Lk, int layer, int causal,
                 float* wout, float* Of, u16* Obf_out) {
    const int Mq = kB * Lq, Mk = kB * Lk;
    const u16* Wl = w_in_bf + (size_t)layer * 3 * kD * kD;
    const float* Bl = b_in + (size_t)layer * 3 * kD;
    gemm(Xq,  Wl,              Bl,          nullptr, Qbf, Mq, kD, kD, 0, 1, 0, 0, 0);
    gemm(Xkv, Wl + kD * kD,    Bl + kD,     nullptr, Kbf, Mk, kD, kD, 0, 1, 0, 0, 0);
    gemm(Xkv, Wl + 2 * kD * kD, Bl + 2 * kD, nullptr, Vbf, Mk, kD, kD, 0, 1, 0, 0, 0);
    int Lkp = (Lk + 31) & ~31;
    k_vt<<<gl((long)kB * kD * Lkp), 256, 0, stream>>>(Vbf, Vt_bf, Lk, Lkp);
    size_t sh = 16 * 512 * 4 + 16 * 512 * 2 + 4 * 16 * 32 * 4 + (wout ? 16 * 512 * 4 : 0);
    dim3 ga((unsigned)((Lq + 15) / 16), (unsigned)kB);
    k_attn<<<ga, 128, sh, stream>>>(Qbf, Kbf, Vt_bf, Obf, wout, Lq, Lk, Lkp, causal,
                                    0.17677669529663687f);
    gemm(Obf, w_out_bf + (size_t)layer * kD * kD, b_out + (size_t)layer * kD, Of,
         Obf_out, Mq, kD, kD, 0, 1, 0, 0, 0);
  };

  // --- weight conversions ---
  k_f2bf<<<gl((long)6 * 3 * kD * kD), 256, 0, stream>>>(w_in, w_in_bf, (long)6 * 3 * kD * kD);
  k_f2bf<<<gl((long)6 * kD * kD), 256, 0, stream>>>(w_out, w_out_bf, (long)6 * kD * kD);
  k_f2bf<<<gl((long)kD * kNumQ), 256, 0, stream>>>(spw, spw_bf, (long)kD * kNumQ);
  k_f2bf<<<gl((long)kD * kN), 256, 0, stream>>>(stew, stew_bf, (long)kD * kN);
  k_f2bf<<<gl((long)kD * kD), 256, 0, stream>>>(f1w, f1w_bf, (long)kD * kD);
  k_f2bf<<<gl((long)kD * kD), 256, 0, stream>>>(f2w, f2w_bf, (long)kD * kD);

  // --- skill table: SWP = softmax(SCW) @ spw^T + spb (2048 x 256) ---
  k_rowsoftmax_bf<<<kNumQ, 256, 0, stream>>>(scw, sw_bf, kNumQ);
  gemm(sw_bf, spw_bf, spb, SWPf, nullptr, kNumQ, kD, kNumQ, 0, 1, 0, 0, 0);

  // --- embeddings ---
  long totMD = kBN * kD;
  k_embed<<<gl(totMD), 256, 0, stream>>>(qi, ri, qryi, Memb, Eemb, Pemb, Mf, Ef, Mbf, Ebf);

  // --- attn0 (E -> M,M, causal, record head-mean weights) ---
  mha(Ebf, kN, Mbf, kN, 0, 1, out + kBN, S0f, nullptr);
  k_add3_ln<<<(unsigned)kBN, 256, 0, stream>>>(S0f, Mf, Ef, ln1g, ln1b, Sbasef, Sbase_bf);

  // --- local attention on last LW rows ---
  k_gather_loc<<<gl((long)kB * kLW * kD), 256, 0, stream>>>(Sbase_bf, loc_bf);
  mha(loc_bf, kLW, loc_bf, kLW, 1, 1, nullptr, Slocout_f, Slocout_bf);
  k_zero16<<<gl(totMD), 256, 0, stream>>>(Slocfull_bf, totMD);
  k_scatter_loc<<<gl((long)kB * kLW * kD), 256, 0, stream>>>(Slocout_bf, Slocfull_bf);

  // --- skill-group path ---
  k_gather_sgr<<<gl(totMD), 256, 0, stream>>>(SWPf, qi, SGR_bf);
  // G[b] = Sbase[b] (N x D) @ SGR[b]^T  -> (N x N), batched
  gemm(Sbase_bf, SGR_bf, nullptr, nullptr, G_bf, kN, kN, kD, 0, kB,
       (long)kN * kD, (long)kN * kD, (long)kN * kN);
  // S_sg0 = G @ stew^T + steb
  gemm(G_bf, stew_bf, steb, nullptr, Ssg0_bf, (int)kBN, kD, kN, 0, 1, 0, 0, 0);
  mha(Ssg0_bf, kN, Ssg0_bf, kN, 2, 0, nullptr, Ssgf, Ssg_bf);

  // --- global / base-local / sg-global attentions ---
  mha(Sbase_bf, kN, Sbase_bf, kN, 3, 1, nullptr, Sglobf, Sglob_bf);
  mha(Sbase_bf, kN, Slocfull_bf, kN, 4, 0, nullptr, Sblf, nullptr);
  mha(Ssg_bf, kN, Sglob_bf, kN, 5, 0, nullptr, Ssglf, nullptr);

  // --- sum + FFN + ln2 + head ---
  k_final_sum<<<gl(totMD), 256, 0, stream>>>(Sbasef, Ssgf, Sglobf, Sblf, Ssglf,
                                             Slocout_f, Ssumf, Ssum_bf);
  gemm(Ssum_bf, f1w_bf, f1b, nullptr, H1_bf, (int)kBN, kD, kD, 1, 1, 0, 0, 0);
  gemm(H1_bf, f2w_bf, f2b, Ff, nullptr, (int)kBN, kD, kD, 0, 1, 0, 0, 0);
  k_add3_ln<<<(unsigned)kBN, 256, 0, stream>>>(Ff, Ssumf, nullptr, ln2g, ln2b, F2f, nullptr);
  k_pred<<<(unsigned)(kBN / 8), 256, 0, stream>>>(F2f, pw, pb, out, (int)kBN);
}
